// PFNet7_3375844295377
// MI455X (gfx1250) — matrix operand order repaired and checked by reading, not compiled, not run on hardware
//
#include <hip/hip_runtime.h>
#include <hip/hip_bf16.h>

typedef __attribute__((ext_vector_type(16))) _Float16 v16h;
typedef __attribute__((ext_vector_type(8)))  float    v8f;

#define NN      6144
#define RAD2    0.01f
#define SLOPE   0.01f
#define HSTRIDE 40   // padded LDS row stride (halfs) for the transposed h chunk

__device__ __forceinline__ float leaky(float v) { return v >= 0.f ? v : SLOPE * v; }

union FragU { uint4 u[2]; v16h v; };

// ---------------------------------------------------------------------------
// Kernel 1: s = x@sW+sb (f32), h = x@hW+hb stored as fp16 rows of 32.
// ---------------------------------------------------------------------------
__global__ __launch_bounds__(256) void pf_embed(
    const float* __restrict__ x, const float* __restrict__ sW, const float* __restrict__ sb,
    const float* __restrict__ hW, const float* __restrict__ hb,
    float* __restrict__ s_out, _Float16* __restrict__ h16_out)
{
    int r = blockIdx.x * 256 + threadIdx.x;
    if (r >= NN) return;
    float xr[12];
    #pragma unroll
    for (int k = 0; k < 12; ++k) xr[k] = x[(size_t)r * 12 + k];

    float s0 = sb[0], s1 = sb[1];
    #pragma unroll
    for (int k = 0; k < 12; ++k) { s0 += xr[k] * sW[k * 2 + 0]; s1 += xr[k] * sW[k * 2 + 1]; }
    s_out[(size_t)r * 2 + 0] = s0;
    s_out[(size_t)r * 2 + 1] = s1;

    _Float16 hrow[32];
    #pragma unroll
    for (int o = 0; o < 32; ++o) {
        float acc = hb[o];
        #pragma unroll
        for (int k = 0; k < 12; ++k) acc += xr[k] * hW[k * 32 + o];
        hrow[o] = (_Float16)acc;
    }
    uint4* dst = (uint4*)(h16_out + (size_t)r * 32);
    const uint4* src = (const uint4*)hrow;
    #pragma unroll
    for (int q = 0; q < 4; ++q) dst[q] = src[q];
}

// ---------------------------------------------------------------------------
// Kernel 2: pairwise radius graph. One block = 16-row i-tile, 8 waves split
// the 6144 j's into 32-wide chunks (24 chunks/wave). h chunk is staged into
// LDS TRANSPOSED ([p][j], padded stride) so that:
//   - each B fragment is two contiguous ds_load_b128
//   - the masked-max loop is two ds_load_b128 per feature p
// wgt is computed straight into the WMMA A fragment; two
// v_wmma_f32_16x16x32_f16 accumulate wgt@h; masked-max uses
// fmaf(w, h, wneg) so masked pairs contribute -3e38 with no cndmask.
// ---------------------------------------------------------------------------
__global__ __launch_bounds__(256) void pf_pairwise(
    const float* __restrict__ s,        // [N][2]
    const _Float16* __restrict__ h16,   // [N][32]
    float* __restrict__ mean_out,       // [N][32]
    float* __restrict__ max_out,        // [N][32]
    float* __restrict__ mask_out)       // [N][N]
{
    __shared__ float     s_i[16][2];
    __shared__ float     sj_sh[8][32][2];
    __shared__ alignas(16) _Float16 hb[8][32][HSTRIDE];  // [wave][p][j] transposed
    __shared__ float     mean_num[16][32];
    __shared__ int       cnt_sh[16];
    __shared__ float     wmax[8][16][32];

    const int tid  = threadIdx.x;
    const int wave = tid >> 5;
    const int lane = tid & 31;
    const int i0   = blockIdx.x * 16;
    const int m    = lane & 15;
    const int i    = i0 + m;

    if (tid < 512) ((float*)mean_num)[tid] = 0.f;
    if (tid < 16)  cnt_sh[tid] = 0;
    if (tid < 32)  ((float*)s_i)[tid] = s[(size_t)i0 * 2 + tid];
    __syncthreads();

    const float sx = s_i[m][0];
    const float sy = s_i[m][1];

    v8f c0 = {}; v8f c1 = {};
    float maxv[32];
    #pragma unroll
    for (int p = 0; p < 32; ++p) maxv[p] = -1e30f;
    int cnt_loc = 0;

    const int base = (lane < 16) ? 0 : 8;   // lane's first K run start
    const int col  = lane & 15;             // B-fragment column
    const int kofs = (lane < 16) ? 0 : 8;   // B-fragment K offset

    // every wave runs exactly 24 iterations -> __syncthreads is uniform
    for (int t = wave; t < NN / 32; t += 8) {
        const int j0 = t * 32;

        // stage s[j] chunk (2 floats/lane)
        sj_sh[wave][lane][0] = s[(size_t)(j0 + lane) * 2 + 0];
        sj_sh[wave][lane][1] = s[(size_t)(j0 + lane) * 2 + 1];
        // stage h chunk transposed: lane's uint4 = 8 halfs of row j, cols po..po+7
        {
            const uint4* hsrc = (const uint4*)(h16 + (size_t)j0 * 32);
            #pragma unroll
            for (int q = 0; q < 4; ++q) {
                const int f  = lane + 32 * q;
                const int j  = f >> 2;
                const int po = (f & 3) * 8;
                uint4 d = hsrc[f];
                const _Float16* hv = (const _Float16*)&d;
                #pragma unroll
                for (int tt = 0; tt < 8; ++tt) hb[wave][po + tt][j] = hv[tt];
            }
        }
        if (j0 + 256 < NN)  // prefetch this wave's next chunk
            __builtin_prefetch(h16 + (size_t)(j0 + 256) * 32, 0, 1);
        __syncthreads();

        // wgt for my 16 (row m, j) pairs, packed per ISA A-fragment layout:
        // lane<16: K = {0..7, 16..23}; lane>=16: K = {8..15, 24..31}
        v16h a;
        float wf[16], wneg[16], mf[16];
        int has = 0;
        #pragma unroll
        for (int e = 0; e < 16; ++e) {
            const int k = (e < 8 ? e : e + 8) + base;
            const int j = j0 + k;
            const float dx = sx - sj_sh[wave][k][0];
            const float dy = sy - sj_sh[wave][k][1];
            const float d2 = dx * dx + dy * dy;
            const bool msk = (d2 <= RAD2) & (j != i);
            const float w = msk ? __expf(-10.f * d2) : 0.f;
            wf[e]   = w;
            wneg[e] = msk ? 0.f : -3e38f;   // folds mask into fmaf for the max
            mf[e]   = msk ? 1.f : 0.f;
            cnt_loc += msk ? 1 : 0;
            has |= (int)msk;
            a[e] = (_Float16)w;
        }

        // mask output: two contiguous 8-float runs per lane -> 4x float4
        {
            const size_t rowbase = (size_t)i * NN;
            const int jb = j0 + base;
            float4* p0 = (float4*)(mask_out + rowbase + jb);
            float4* p1 = (float4*)(mask_out + rowbase + jb + 16);
            p0[0] = make_float4(mf[0],  mf[1],  mf[2],  mf[3]);
            p0[1] = make_float4(mf[4],  mf[5],  mf[6],  mf[7]);
            p1[0] = make_float4(mf[8],  mf[9],  mf[10], mf[11]);
            p1[1] = make_float4(mf[12], mf[13], mf[14], mf[15]);
        }

        if (__ballot(has) != 0ull) {   // wave-uniform: EXEC all-ones for WMMA
            // B fragments: lane<16 -> col=lane, K={0..7,16..23}; lane>=16 -> col-16, K={8..15,24..31}.
            // Transposed LDS makes each fragment two contiguous b128 loads.
            FragU f0, f1;
            const uint4* q0 = (const uint4*)&hb[wave][col][kofs];
            const uint4* q1 = (const uint4*)&hb[wave][col + 16][kofs];
            f0.u[0] = q0[0]; f0.u[1] = q0[2];
            f1.u[0] = q1[0]; f1.u[1] = q1[2];
            c0 = __builtin_amdgcn_wmma_f32_16x16x32_f16(
                     false, a, false, f0.v, (short)0, c0, false, false);
            c1 = __builtin_amdgcn_wmma_f32_16x16x32_f16(
                     false, a, false, f1.v, (short)0, c1, false, false);

            // masked max of wgt*h: p-outer, two b128 loads of lane's K runs per p.
            // FULLY unrolled so maxv[] indices stay static (no v_movreld).
            #pragma unroll
            for (int p = 0; p < 32; ++p) {
                const uint4 r1 = *(const uint4*)&hb[wave][p][base];       // K base..base+7
                const uint4 r2 = *(const uint4*)&hb[wave][p][base + 16];  // K base+16..+23
                const _Float16* h1 = (const _Float16*)&r1;
                const _Float16* h2 = (const _Float16*)&r2;
                float mv = maxv[p];
                #pragma unroll
                for (int e = 0; e < 8; ++e)
                    mv = fmaxf(mv, fmaf(wf[e], (float)h1[e], wneg[e]));
                #pragma unroll
                for (int e = 0; e < 8; ++e)
                    mv = fmaxf(mv, fmaf(wf[e + 8], (float)h2[e], wneg[e + 8]));
                maxv[p] = mv;
            }
        }
        __syncthreads();
    }

    // cross-wave reductions
    atomicAdd(&cnt_sh[m], cnt_loc);
    #pragma unroll
    for (int r = 0; r < 8; ++r) {
        const int mm = (lane < 16) ? r : (8 + r);   // WMMA C layout
        const int n  = lane & 15;
        atomicAdd(&mean_num[mm][n],      c0[r]);
        atomicAdd(&mean_num[mm][n + 16], c1[r]);
    }
    #pragma unroll
    for (int p = 0; p < 32; ++p) {
        float v = maxv[p];
        v = fmaxf(v, __shfl_xor(v, 16, 32));        // merge lane m / m+16
        if (lane < 16) wmax[wave][m][p] = v;
    }
    __syncthreads();

    for (int idx = tid; idx < 512; idx += 256) {
        const int mm = idx >> 5, p = idx & 31;
        float mx = wmax[0][mm][p];
        #pragma unroll
        for (int w = 1; w < 8; ++w) mx = fmaxf(mx, wmax[w][mm][p]);
        const int c = cnt_sh[mm];
        const float inv = 1.f / (float)(c > 1 ? c : 1);
        const size_t o = (size_t)(i0 + mm) * 32 + p;
        mean_out[o] = mean_num[mm][p] * inv;
        max_out[o]  = (c > 0) ? mx : 0.f;
    }
}

// ---------------------------------------------------------------------------
// Kernel 3: encoder + both MLP heads. One wave per node (8 waves/block).
// ---------------------------------------------------------------------------
__global__ __launch_bounds__(256) void pf_heads(
    const float* __restrict__ x, const float* __restrict__ mean_a, const float* __restrict__ max_a,
    const float* __restrict__ oW, const float* __restrict__ ob,
    const float* __restrict__ W1, const float* __restrict__ b1,
    const float* __restrict__ W2, const float* __restrict__ b2,
    const float* __restrict__ W3, const float* __restrict__ b3,
    const float* __restrict__ W4, const float* __restrict__ b4,
    const float* __restrict__ V1, const float* __restrict__ d1,
    const float* __restrict__ V2, const float* __restrict__ d2,
    const float* __restrict__ V3, const float* __restrict__ d3,
    const float* __restrict__ V4, const float* __restrict__ d4,
    float* __restrict__ out_ids, float* __restrict__ out_p4)
{
    __shared__ float feat[8][80];
    __shared__ float x1b[8][256];
    __shared__ float idb[8][8];

    const int wave = threadIdx.x >> 5;
    const int lane = threadIdx.x & 31;
    const int r = blockIdx.x * 8 + wave;

    if (lane < 12) feat[wave][lane] = x[(size_t)r * 12 + lane];
    feat[wave][12 + lane] = mean_a[(size_t)r * 32 + lane];
    feat[wave][44 + lane] = max_a[(size_t)r * 32 + lane];
    __syncthreads();

    // encoder: 76 -> 256, leaky
    #pragma unroll
    for (int u = 0; u < 8; ++u) {
        const int c = lane + 32 * u;
        float acc = ob[c];
        for (int k = 0; k < 76; ++k) acc += feat[wave][k] * oW[k * 256 + c];
        x1b[wave][c] = leaky(acc);
    }
    __syncthreads();

    // nn2: 256 -> 32 -> 32 -> 32 -> 6
    float v = b1[lane];
    for (int k = 0; k < 256; ++k) v += x1b[wave][k] * W1[k * 32 + lane];
    v = leaky(v);
    float v2 = b2[lane];
    #pragma unroll
    for (int k = 0; k < 32; ++k) v2 += __shfl(v, k, 32) * W2[k * 32 + lane];
    v2 = leaky(v2);
    float v3 = b3[lane];
    #pragma unroll
    for (int k = 0; k < 32; ++k) v3 += __shfl(v2, k, 32) * W3[k * 32 + lane];
    v3 = leaky(v3);
    {
        const int o = (lane < 6) ? lane : 0;
        float acc = b4[o];
        #pragma unroll
        for (int k = 0; k < 32; ++k) acc += __shfl(v3, k, 32) * W4[k * 6 + o];
        if (lane < 6) {
            idb[wave][lane] = acc;
            out_ids[(size_t)r * 6 + lane] = acc;
        }
    }
    __syncthreads();

    // nn3: 262 -> 32 -> 32 -> 32 -> 6
    float u1 = d1[lane];
    for (int k = 0; k < 256; ++k) u1 += x1b[wave][k] * V1[k * 32 + lane];
    #pragma unroll
    for (int k = 0; k < 6; ++k) u1 += idb[wave][k] * V1[(256 + k) * 32 + lane];
    u1 = leaky(u1);
    float u2 = d2[lane];
    #pragma unroll
    for (int k = 0; k < 32; ++k) u2 += __shfl(u1, k, 32) * V2[k * 32 + lane];
    u2 = leaky(u2);
    float u3 = d3[lane];
    #pragma unroll
    for (int k = 0; k < 32; ++k) u3 += __shfl(u2, k, 32) * V3[k * 32 + lane];
    u3 = leaky(u3);
    {
        const int o = (lane < 6) ? lane : 0;
        float acc = d4[o];
        #pragma unroll
        for (int k = 0; k < 32; ++k) acc += __shfl(u3, k, 32) * V4[k * 6 + o];
        if (lane < 6) out_p4[(size_t)r * 6 + lane] = acc;
    }
}

// ---------------------------------------------------------------------------
extern "C" void kernel_launch(void* const* d_in, const int* in_sizes, int n_in,
                              void* d_out, int out_size, void* d_ws, size_t ws_size,
                              hipStream_t stream) {
    (void)in_sizes; (void)n_in; (void)out_size; (void)ws_size;

    const float* x  = (const float*)d_in[0];
    const float* sW = (const float*)d_in[1];
    const float* sb = (const float*)d_in[2];
    const float* hW = (const float*)d_in[3];
    const float* hb = (const float*)d_in[4];
    const float* oW = (const float*)d_in[5];
    const float* ob = (const float*)d_in[6];
    // nn2: (W,b) x4 ; nn3: (W,b) x4
    const float* W1 = (const float*)d_in[7];  const float* b1 = (const float*)d_in[8];
    const float* W2 = (const float*)d_in[9];  const float* b2 = (const float*)d_in[10];
    const float* W3 = (const float*)d_in[11]; const float* b3 = (const float*)d_in[12];
    const float* W4 = (const float*)d_in[13]; const float* b4 = (const float*)d_in[14];
    const float* V1 = (const float*)d_in[15]; const float* d1 = (const float*)d_in[16];
    const float* V2 = (const float*)d_in[17]; const float* d2 = (const float*)d_in[18];
    const float* V3 = (const float*)d_in[19]; const float* d3 = (const float*)d_in[20];
    const float* V4 = (const float*)d_in[21]; const float* d4 = (const float*)d_in[22];

    // workspace layout (all 16B aligned)
    char* ws = (char*)d_ws;
    float*    s_ws    = (float*)(ws);                         // 6144*2*4   = 48 KB
    _Float16* h16_ws  = (_Float16*)(ws + 49152);              // 6144*32*2  = 384 KB
    float*    mean_ws = (float*)(ws + 49152 + 393216);        // 6144*32*4  = 768 KB
    float*    max_ws  = (float*)(ws + 49152 + 393216 + 786432);

    float* out = (float*)d_out;
    float* out_ids  = out;                        // [N][6]
    float* out_p4   = out + (size_t)NN * 6;       // [N][6]
    float* out_mask = out + (size_t)NN * 12;      // [N][N]

    pf_embed<<<NN / 256, 256, 0, stream>>>(x, sW, sb, hW, hb, s_ws, h16_ws);
    pf_pairwise<<<NN / 16, 256, 0, stream>>>(s_ws, h16_ws, mean_ws, max_ws, out_mask);
    pf_heads<<<NN / 8, 256, 0, stream>>>(x, mean_ws, max_ws, oW, ob,
                                         W1, b1, W2, b2, W3, b3, W4, b4,
                                         V1, d1, V2, d2, V3, d3, V4, d4,
                                         out_ids, out_p4);
}
